// FoldedQuantizedResNet_30623116821394
// MI455X (gfx1250) — 8-bit matrix operand order repaired and checked by reading, compiled, not run on hardware
//
#include <hip/hip_runtime.h>

typedef __attribute__((ext_vector_type(8))) int v8i;


__device__ __constant__ const long long kM0Q = 2097;
#define K_SHIFT 20
#define K_RND   (1LL << (K_SHIFT - 1))

__global__ void quantize_kernel(const float* __restrict__ x,
                                unsigned char* __restrict__ q, long long n) {
  long long i = (long long)blockIdx.x * blockDim.x + threadIdx.x;
  if (i >= n) return;
  float r = rintf(x[i] * 50.0f) + 128.0f;
  r = fminf(fmaxf(r, 0.0f), 255.0f);
  q[i] = (unsigned char)(int)r;
}

__global__ void repack_a_kernel(const int* __restrict__ w,
                                unsigned int* __restrict__ frag,
                                int rows, int Kreal, int rT, int KC) {
  long long tid = (long long)blockIdx.x * blockDim.x + threadIdx.x;
  long long total = (long long)rT * KC * 256;
  if (tid >= total) return;
  int v = (int)(tid & 7);
  int lane = (int)((tid >> 3) & 31);
  long long ct = tid >> 8;
  int kc = (int)(ct % KC);
  int r_t = (int)(ct / KC);
  int row = r_t * 16 + (lane & 15);
  int khi = (lane & 16) ? 16 : 0;
  unsigned int word = 0;
  for (int b = 0; b < 4; ++b) {
    int kl = ((v >> 2) << 5) + khi + ((v & 3) << 2) + b;
    int K = kc * 64 + kl;
    int val = 0;
    if (row < rows && K < Kreal) val = w[(long long)row * Kreal + K];
    word |= ((unsigned int)(val & 0xFF)) << (8 * b);
  }
  frag[tid] = word;
}

__global__ void repack_b_kernel(const int* __restrict__ w,
                                unsigned int* __restrict__ frag,
                                int cols, int Kreal, int nT, int KC) {
  long long tid = (long long)blockIdx.x * blockDim.x + threadIdx.x;
  long long total = (long long)nT * KC * 256;
  if (tid >= total) return;
  int v = (int)(tid & 7);
  int lane = (int)((tid >> 3) & 31);
  long long ct = tid >> 8;
  int kc = (int)(ct % KC);
  long long n_t = ct / KC;
  int col = (int)(n_t * 16 + (lane & 15));
  int khi = (lane & 16) ? 16 : 0;
  unsigned int word = 0;
  for (int b = 0; b < 4; ++b) {
    int kl = ((v >> 2) << 5) + khi + ((v & 3) << 2) + b;
    int K = kc * 64 + kl;
    int val = 0;
    if (col < cols && K < Kreal) val = w[(long long)col * Kreal + K];
    word |= ((unsigned int)(val & 0xFF)) << (8 * b);
  }
  frag[tid] = word;
}

__global__ void im2col_kernel(const unsigned char* __restrict__ in,
                              unsigned int* __restrict__ frag,
                              int cin, int IH, int IW, int ks, int stride,
                              int pad, int OH, int OW, int zp, int KC,
                              long long nT) {
  long long tid = (long long)blockIdx.x * blockDim.x + threadIdx.x;
  long long total = nT * KC * 256;
  if (tid >= total) return;
  int v = (int)(tid & 7);
  int lane = (int)((tid >> 3) & 31);
  long long ct = tid >> 8;
  int kc = (int)(ct % KC);
  long long n_t = ct / KC;
  long long OHW = (long long)OH * OW;
  long long Ncols = 32LL * OHW;
  long long col = n_t * 16 + (lane & 15);
  int ksq = ks * ks;
  int Kreal = cin * ksq;
  int khi = (lane & 16) ? 16 : 0;
  unsigned int word = 0;
  if (col < Ncols) {
    int nimg = (int)(col / OHW);
    int rem = (int)(col % OHW);
    int oh = rem / OW, ow = rem % OW;
    for (int b = 0; b < 4; ++b) {
      int kl = ((v >> 2) << 5) + khi + ((v & 3) << 2) + b;
      int K = kc * 64 + kl;
      int val = 0;
      if (K < Kreal) {
        int ci = K / ksq;
        int r = K % ksq;
        int kh = r / ks, kw = r % ks;
        int ih = oh * stride - pad + kh;
        int iw = ow * stride - pad + kw;
        if (ih >= 0 && ih < IH && iw >= 0 && iw < IW)
          val = (int)in[(((long long)nimg * cin + ci) * IH + ih) * IW + iw] - zp;
      }
      word |= ((unsigned int)(val & 0xFF)) << (8 * b);
    }
  }
  frag[tid] = word;
}

template <bool BSGN>
__global__ void __launch_bounds__(256)
conv_gemm_wmma(const unsigned char* __restrict__ Af,
               const unsigned char* __restrict__ Bf,
               unsigned char* __restrict__ out,
               const unsigned char* __restrict__ residual,
               int coT2, long long nT4, int KC, int cout, int OHW,
               long long Ncols) {
  long long wave = (long long)blockIdx.x * 8 + (threadIdx.x >> 5);
  long long total = (long long)coT2 * nT4;
  if (wave >= total) return;
  int lane = threadIdx.x & 31;
  int co2 = (int)(wave % coT2);
  long long nt4 = wave / coT2;

  const unsigned char* apb[2];
  const unsigned char* bpb[4];
#pragma unroll
  for (int i = 0; i < 2; ++i)
    apb[i] = Af + ((long long)(co2 * 2 + i) * KC) * 1024 + lane * 32;
#pragma unroll
  for (int j = 0; j < 4; ++j)
    bpb[j] = Bf + ((nt4 * 4 + j) * KC) * 1024 + lane * 32;

  v8i acc[2][4];
#pragma unroll
  for (int i = 0; i < 2; ++i)
#pragma unroll
    for (int j = 0; j < 4; ++j) acc[i][j] = (v8i){0, 0, 0, 0, 0, 0, 0, 0};

  for (int kc = 0; kc < KC; ++kc) {
    v8i a[2], b[4];
#pragma unroll
    for (int i = 0; i < 2; ++i)
      a[i] = *(const v8i*)(apb[i] + (long long)kc * 1024);
#pragma unroll
    for (int j = 0; j < 4; ++j)
      b[j] = *(const v8i*)(bpb[j] + (long long)kc * 1024);
    if (kc + 1 < KC) {
#pragma unroll
      for (int i = 0; i < 2; ++i)
        __builtin_prefetch(apb[i] + (long long)(kc + 1) * 1024, 0, 1);
#pragma unroll
      for (int j = 0; j < 4; ++j)
        __builtin_prefetch(bpb[j] + (long long)(kc + 1) * 1024, 0, 1);
    }
#pragma unroll
    for (int i = 0; i < 2; ++i)
#pragma unroll
      for (int j = 0; j < 4; ++j)
        acc[i][j] = __builtin_amdgcn_wmma_i32_16x16x64_iu8(
            true, a[i], BSGN, b[j], acc[i][j], false, false);
  }

  int colL = lane & 15;
  int rowAdd = (lane & 16) ? 8 : 0;
#pragma unroll
  for (int j = 0; j < 4; ++j) {
    long long col = (nt4 * 4 + j) * 16 + colL;
    if (col >= Ncols) continue;
    int nimg = (int)(col / OHW);
    int rem = (int)(col % OHW);
#pragma unroll
    for (int i = 0; i < 2; ++i) {
      int coBase = (co2 * 2 + i) * 16 + rowAdd;
#pragma unroll
      for (int k = 0; k < 8; ++k) {
        int co = coBase + k;
        if (co >= cout) continue;
        long long a64 = (long long)acc[i][j][k] * kM0Q;
        long long r = (a64 + K_RND) >> K_SHIFT;
        long long oi = ((long long)nimg * cout + co) * OHW + rem;
        if (residual) r += (long long)residual[oi];
        r = r < 0 ? 0 : (r > 255 ? 255 : r);
        out[oi] = (unsigned char)r;
      }
    }
  }
}

__global__ void maxpool_kernel(const unsigned char* __restrict__ in,
                               unsigned char* __restrict__ out,
                               int NC, int IH, int IW, int OH, int OW) {
  long long tid = (long long)blockIdx.x * blockDim.x + threadIdx.x;
  long long total = (long long)NC * OH * OW;
  if (tid >= total) return;
  int ow = (int)(tid % OW);
  int oh = (int)((tid / OW) % OH);
  long long nc = tid / ((long long)OH * OW);
  const unsigned char* base = in + nc * (long long)IH * IW;
  int best = 0;
  for (int kh = 0; kh < 3; ++kh) {
    int ih = oh * 2 - 1 + kh;
    if (ih < 0 || ih >= IH) continue;
    for (int kw = 0; kw < 3; ++kw) {
      int iw = ow * 2 - 1 + kw;
      if (iw < 0 || iw >= IW) continue;
      int vv = base[ih * IW + iw];
      if (vv > best) best = vv;
    }
  }
  out[tid] = (unsigned char)best;
}

__global__ void avgpool_requant_kernel(const unsigned char* __restrict__ in,
                                       int* __restrict__ t, int total, int HW) {
  int tid = blockIdx.x * blockDim.x + threadIdx.x;
  if (tid >= total) return;
  const unsigned char* p = in + (long long)tid * HW;
  int sum = 0;
  for (int i = 0; i < HW; ++i) sum += p[i];
  float f = truncf((float)sum / (float)HW);
  int ti = (int)f;
  ti = ti < 0 ? 0 : (ti > 255 ? 255 : ti);
  t[tid] = ti;
}

__global__ void __launch_bounds__(256)
fc_gemm_wmma(const unsigned char* __restrict__ Af,
             const unsigned char* __restrict__ Bf,
             const int* __restrict__ bias, float* __restrict__ out,
             int mT, int nT, int KC, int Mreal, int Nreal) {
  int wave = blockIdx.x * 8 + (threadIdx.x >> 5);
  if (wave >= mT * nT) return;
  int lane = threadIdx.x & 31;
  int m_t = wave % mT;
  int n_t = wave / mT;
  const unsigned char* ap = Af + ((long long)m_t * KC) * 1024 + lane * 32;
  const unsigned char* bp = Bf + ((long long)n_t * KC) * 1024 + lane * 32;
  v8i c = {0, 0, 0, 0, 0, 0, 0, 0};
  for (int kc = 0; kc < KC; ++kc) {
    v8i a = *(const v8i*)(ap + (long long)kc * 1024);
    v8i b = *(const v8i*)(bp + (long long)kc * 1024);
    c = __builtin_amdgcn_wmma_i32_16x16x64_iu8(false, a, true, b, c,
                                               false, false);
  }
  int col = n_t * 16 + (lane & 15);
  int rowAdd = (lane & 16) ? 8 : 0;
  if (col >= Nreal) return;
  int bi = bias[col];
  for (int i = 0; i < 8; ++i) {
    int row = m_t * 16 + i + rowAdd;
    if (row < Mreal) out[(long long)row * Nreal + col] = (float)(c[i] + bi);
  }
}

extern "C" void kernel_launch(void* const* d_in, const int* in_sizes, int n_in,
                              void* d_out, int out_size, void* d_ws,
                              size_t ws_size, hipStream_t stream) {
  (void)in_sizes; (void)n_in; (void)out_size; (void)ws_size;

  static const int  cinA[8]  = {64, 64, 64, 128, 128, 256, 256, 512};
  static const int  coutA[8] = {64, 64, 128, 128, 256, 256, 512, 512};
  static const int  strA[8]  = {1, 1, 2, 1, 2, 1, 2, 1};
  static const bool downA[8] = {false, false, true, false, true, false, true, false};

  const float* x = (const float*)d_in[0];
  const int* w_first = (const int*)d_in[1];
  const int* w1[8]; const int* w2[8]; const int* wd[8];
  int idx = 2;
  for (int b = 0; b < 8; ++b) {
    w1[b] = (const int*)d_in[idx++];
    w2[b] = (const int*)d_in[idx++];
    wd[b] = downA[b] ? (const int*)d_in[idx++] : nullptr;
  }
  const int* w_fc = (const int*)d_in[idx++];
  const int* b_fc = (const int*)d_in[idx++];
  float* out = (float*)d_out;

  const int B = 32;
  char* ws = (char*)d_ws;
  size_t off = 0;
  auto alloc = [&](size_t bytes) -> void* {
    void* p = ws + off;
    off = (off + bytes + 255) & ~(size_t)255;
    return p;
  };
  auto fragBytesA = [](int rows, int Kreal) -> size_t {
    return (size_t)((rows + 15) / 16) * ((Kreal + 63) / 64) * 1024;
  };

  unsigned char* qin    = (unsigned char*)alloc((size_t)B * 3 * 224 * 224);
  unsigned char* conv1  = (unsigned char*)alloc((size_t)B * 64 * 112 * 112);
  unsigned char* pooled = (unsigned char*)alloc((size_t)B * 64 * 56 * 56);
  size_t actBytes = (size_t)B * 64 * 56 * 56;
  unsigned char* actbuf[4];
  for (int i = 0; i < 4; ++i) actbuf[i] = (unsigned char*)alloc(actBytes);
  unsigned char* colbuf = (unsigned char*)alloc((size_t)(25088 + 4) * 3 * 1024);

  unsigned char* wf_first = (unsigned char*)alloc(fragBytesA(64, 3 * 49));
  unsigned char* wf1[8]; unsigned char* wf2[8]; unsigned char* wfD[8];
  for (int b = 0; b < 8; ++b) {
    wf1[b] = (unsigned char*)alloc(fragBytesA(coutA[b], cinA[b] * 9));
    wf2[b] = (unsigned char*)alloc(fragBytesA(coutA[b], coutA[b] * 9));
    wfD[b] = downA[b] ? (unsigned char*)alloc(fragBytesA(coutA[b], cinA[b]))
                      : nullptr;
  }
  unsigned char* wf_fc = (unsigned char*)alloc((size_t)63 * 8 * 1024);
  int* tvec            = (int*)alloc((size_t)B * 512 * 4);
  unsigned char* tfrag = (unsigned char*)alloc(fragBytesA(32, 512));

  auto launch_repackA = [&](const int* w, unsigned char* frag, int rows,
                            int Kreal) {
    int rT = (rows + 15) / 16, KC = (Kreal + 63) / 64;
    long long total = (long long)rT * KC * 256;
    int blocks = (int)((total + 255) / 256);
    repack_a_kernel<<<blocks, 256, 0, stream>>>(w, (unsigned int*)frag, rows,
                                                Kreal, rT, KC);
  };
  auto launch_repackB = [&](const int* w, unsigned char* frag, int cols,
                            int Kreal) {
    int nT = (cols + 15) / 16, KC = (Kreal + 63) / 64;
    long long total = (long long)nT * KC * 256;
    int blocks = (int)((total + 255) / 256);
    repack_b_kernel<<<blocks, 256, 0, stream>>>(w, (unsigned int*)frag, cols,
                                                Kreal, nT, KC);
  };
  auto nTilesPadded = [&](int OH, int OW) -> long long {
    long long Ncols = (long long)B * OH * OW;
    long long nT = (Ncols + 15) / 16;
    return ((nT + 3) / 4) * 4;
  };
  auto launch_im2col = [&](const unsigned char* in, int cin, int IH, int IW,
                           int ks, int stride, int pad, int OH, int OW,
                           int zp) {
    int KC = (cin * ks * ks + 63) / 64;
    long long nT = nTilesPadded(OH, OW);
    long long total = nT * KC * 256;
    int blocks = (int)((total + 255) / 256);
    im2col_kernel<<<blocks, 256, 0, stream>>>(in, (unsigned int*)colbuf, cin,
                                              IH, IW, ks, stride, pad, OH, OW,
                                              zp, KC, nT);
  };
  auto launch_gemm = [&](const unsigned char* Af, unsigned char* outp,
                         const unsigned char* resid, int cout, int Kreal,
                         int OH, int OW, bool bsgn) {
    int coT2 = ((cout + 15) / 16) / 2;
    int KC = (Kreal + 63) / 64;
    long long Ncols = (long long)B * OH * OW;
    long long nT4 = nTilesPadded(OH, OW) / 4;
    long long waves = (long long)coT2 * nT4;
    int blocks = (int)((waves + 7) / 8);
    if (bsgn)
      conv_gemm_wmma<true><<<blocks, 256, 0, stream>>>(
          Af, colbuf, outp, resid, coT2, nT4, KC, cout, OH * OW, Ncols);
    else
      conv_gemm_wmma<false><<<blocks, 256, 0, stream>>>(
          Af, colbuf, outp, resid, coT2, nT4, KC, cout, OH * OW, Ncols);
  };

  {
    long long n = (long long)B * 3 * 224 * 224;
    int blocks = (int)((n + 255) / 256);
    quantize_kernel<<<blocks, 256, 0, stream>>>(x, qin, n);
  }
  launch_repackA(w_first, wf_first, 64, 147);
  for (int b = 0; b < 8; ++b) {
    launch_repackA(w1[b], wf1[b], coutA[b], cinA[b] * 9);
    launch_repackA(w2[b], wf2[b], coutA[b], coutA[b] * 9);
    if (downA[b]) launch_repackA(wd[b], wfD[b], coutA[b], cinA[b]);
  }
  launch_repackB(w_fc, wf_fc, 1000, 512);
  launch_im2col(qin, 3, 224, 224, 7, 2, 3, 112, 112, 128);
  launch_gemm(wf_first, conv1, nullptr, 64, 147, 112, 112, true);
  {
    long long total = (long long)B * 64 * 56 * 56;
    int blocks = (int)((total + 255) / 256);
    maxpool_kernel<<<blocks, 256, 0, stream>>>(conv1, pooled, B * 64, 112, 112,
                                               56, 56);
  }
  const unsigned char* cur = pooled;
  int H = 56;
  for (int b = 0; b < 8; ++b) {
    int cin = cinA[b], cout = coutA[b], s = strA[b];
    int OH = H / s;
    unsigned char* scratch[3];
    int si = 0;
    for (int i = 0; i < 4 && si < 3; ++i)
      if (actbuf[i] != (unsigned char*)cur) scratch[si++] = actbuf[i];
    unsigned char* tmp  = scratch[0];
    unsigned char* outb = scratch[1];
    unsigned char* idb  = scratch[2];
    const unsigned char* resid;
    if (downA[b]) {
      launch_im2col(cur, cin, H, H, 1, s, 0, OH, OH, 0);
      launch_gemm(wfD[b], idb, nullptr, cout, cin, OH, OH, false);
      resid = idb;
    } else {
      resid = cur;
    }
    launch_im2col(cur, cin, H, H, 3, s, 1, OH, OH, 0);
    launch_gemm(wf1[b], tmp, nullptr, cout, cin * 9, OH, OH, false);
    launch_im2col(tmp, cout, OH, OH, 3, 1, 1, OH, OH, 0);
    launch_gemm(wf2[b], outb, resid, cout, cout * 9, OH, OH, false);
    cur = outb;
    H = OH;
  }
  {
    int total = B * 512;
    int blocks = (total + 255) / 256;
    avgpool_requant_kernel<<<blocks, 256, 0, stream>>>(cur, tvec, total, 49);
  }
  launch_repackA(tvec, tfrag, 32, 512);
  {
    int waves = 2 * 63;
    int blocks = (waves + 7) / 8;
    fc_gemm_wmma<<<blocks, 256, 0, stream>>>(tfrag, wf_fc, b_fc, out, 2, 63, 8,
                                             32, 1000);
  }
}
